// aRNN_75694503624776
// MI455X (gfx1250) — compile-verified, matrix operand order checked by reading, not run on hardware
//
#include <hip/hip_runtime.h>
#include <hip/hip_bf16.h>

// ---------- types ----------
typedef _Float16 v16h __attribute__((ext_vector_type(16)));
typedef _Float16 v8h  __attribute__((ext_vector_type(8)));
typedef float    v8f  __attribute__((ext_vector_type(8)));
typedef float    v4f  __attribute__((ext_vector_type(4)));

// Problem constants
#define TT   1024
#define BB   64
#define NWG  16           // persistent workgroups (all resident -> safe grid barrier)
// per-layer fragment region sizes (fragments of 512 f16 each)
#define FR_LSTM  1280     // 32 ntiles * 40 ktiles   (K = 1152 + 128)
#define FR_POL   768      // 192 ntiles * 4 ktiles   (K = 128, N = 3072)
#define FR_F     1024     // 32 ntiles * 32 ktiles   (K = 512 + 512)
#define FR_PERL  (FR_LSTM + FR_POL + FR_F)   // 3072 frags / layer

// ---------- global-address-space load/store helpers ----------
// Force addrspace(1) so the backend emits global_load/global_store (LOADcnt
// only) instead of flat_* (which also ties up DScnt and the LDS path).
#define GAS __attribute__((address_space(1)))
static __device__ __forceinline__ v8h  gld8h (const _Float16* p) { return *(const GAS v8h*)p; }
static __device__ __forceinline__ v16h gld16h(const _Float16* p) { return *(const GAS v16h*)p; }
static __device__ __forceinline__ v4f  gld4f (const float* p)    { return *(const GAS v4f*)p; }
static __device__ __forceinline__ float gldf (const float* p)    { return *(const GAS float*)p; }
static __device__ __forceinline__ void gstf (float* p, float v)      { *(GAS float*)p = v; }
static __device__ __forceinline__ void gsth (_Float16* p, _Float16 v){ *(GAS _Float16*)p = v; }

// ---------- WMMA wrapper ----------
static __device__ __forceinline__ v8f wmma16(v16h a, v16h b, v8f c) {
  return __builtin_amdgcn_wmma_f32_16x16x32_f16(false, a, false, b, (short)0, c,
                                                false, false);
}

// A fragment (16x32, f16 source): lane holds row m = lane&15, k-chunks at
// c = kt*32 + (lane>>4)*8  and  c+16  (8 contiguous f16 each, per ISA layout).
static __device__ __forceinline__ v16h afrag_h(const _Float16* src, int stride,
                                               int m, int c) {
  const _Float16* p = src + (size_t)m * stride + c;
  v8h lo = gld8h(p);
  v8h hi = gld8h(p + 16);
  v16h r;
#pragma unroll
  for (int i = 0; i < 8; ++i) { r[i] = lo[i]; r[i + 8] = hi[i]; }
  return r;
}

// A fragment converting from fp32 source on the fly (used for the raw x input)
static __device__ __forceinline__ v16h afrag_f(const float* src, int stride,
                                               int m, int c) {
  const float* p = src + (size_t)m * stride + c;
  v4f a0 = gld4f(p);
  v4f a1 = gld4f(p + 4);
  v4f b0 = gld4f(p + 16);
  v4f b1 = gld4f(p + 20);
  v16h r;
#pragma unroll
  for (int i = 0; i < 4; ++i) {
    r[i]      = (_Float16)a0[i];
    r[4 + i]  = (_Float16)a1[i];
    r[8 + i]  = (_Float16)b0[i];
    r[12 + i] = (_Float16)b1[i];
  }
  return r;
}

// Pre-swizzled weight fragment: 32 bytes per lane, contiguous.
static __device__ __forceinline__ v16h bfrag(const _Float16* wb, int frag, int lane) {
  return gld16h(wb + (size_t)frag * 512 + lane * 16);
}

static __device__ __forceinline__ float sigm(float x) {
  return 1.0f / (1.0f + __expf(-x));
}

// ---------- device-wide sense-reversal barrier (16 resident WGs) ----------
static __device__ __forceinline__ void grid_barrier(unsigned* cnt, unsigned* gen) {
  __syncthreads();
  if (threadIdx.x == 0) {
    unsigned g = __hip_atomic_load(gen, __ATOMIC_RELAXED, __HIP_MEMORY_SCOPE_AGENT);
    unsigned old = __hip_atomic_fetch_add(cnt, 1u, __ATOMIC_ACQ_REL,
                                          __HIP_MEMORY_SCOPE_AGENT);
    if (old == NWG - 1) {
      __hip_atomic_store(cnt, 0u, __ATOMIC_RELAXED, __HIP_MEMORY_SCOPE_AGENT);
      __hip_atomic_store(gen, g + 1u, __ATOMIC_RELEASE, __HIP_MEMORY_SCOPE_AGENT);
    } else {
      while (__hip_atomic_load(gen, __ATOMIC_ACQUIRE, __HIP_MEMORY_SCOPE_AGENT) == g)
        __builtin_amdgcn_s_sleep(1);
    }
  }
  __syncthreads();
}

// ---------- argument bundle ----------
struct Args {
  const float* x;
  const float *bih0, *bhh0, *pb0, *fb0;
  const float *bih1, *bhh1, *pb1, *fb1;
  const _Float16* wfrag;                 // swizzled weights, 2 layers
  _Float16 *ahx0h, *ahx1h, *fhx0h, *fhx1h;   // f16 state mirrors [2 slots]
  float *ahc0, *ahc1, *fhx0f, *fhx1f;        // fp32 states [2 slots]
  unsigned* bar;                         // [0]=count [1]=generation
  float* polbuf;                         // 64 x 3072 fp32 (per-step scratch)
  _Float16* uA;                          // 64 x 1024 f16  (x*p_in | h*p_h)
  float* out;
};

// accumulate one A-fragment against all 4 gate weight tiles
static __device__ __forceinline__ void gate4(v8f acc[4], v16h a,
                                             const _Float16* wb, int ntb,
                                             int ktglob, int lane) {
#pragma unroll
  for (int g = 0; g < 4; ++g)
    acc[g] = wmma16(a, bfrag(wb, (ntb + 8 * g) * 40 + ktglob, lane), acc[g]);
}

// ---------- phase: fused LSTM (GEMM K=1280 + in-register gate epilogue) ----
// waves 0..31 each own 16 units x 16 batch rows across all 4 gates.
static __device__ void lstm_phase(int wave, int lane,
    const float* xf, const _Float16* xh,      // region0 (K 0..511): fp32 or f16
    const _Float16* hp,                       // region1 (K 512..1023), stride 512
    const _Float16* aoth,                     // region2 (K 1024..1151), stride 128
    const _Float16* aself,                    // region3 (K 1152..1279), stride 128
    const _Float16* wb, const float* bih, const float* bhh,
    const float* ahcs, float* ahcd, _Float16* ahxd) {
  if (wave >= 32) return;
  int mt = wave >> 3, ntb = wave & 7;
  int mrow = mt * 16 + (lane & 15);
  int ch   = (lane >> 4) * 8;
  v8f acc[4] = {};
  // region 0: x role (fp32 raw input for layer0, f16 mirror for layer1)
  if (xf) {
#pragma unroll 2
    for (int kt = 0; kt < 16; ++kt)
      gate4(acc, afrag_f(xf, 512, mrow, kt * 32 + ch), wb, ntb, kt, lane);
  } else {
#pragma unroll 2
    for (int kt = 0; kt < 16; ++kt)
      gate4(acc, afrag_h(xh, 512, mrow, kt * 32 + ch), wb, ntb, kt, lane);
  }
  // region 1: h role (prev fhx of this layer)
#pragma unroll 2
  for (int kt = 0; kt < 16; ++kt)
    gate4(acc, afrag_h(hp, 512, mrow, kt * 32 + ch), wb, ntb, 16 + kt, lane);
  // region 2: other controller state (ASZ = 128)
#pragma unroll
  for (int kt = 0; kt < 4; ++kt)
    gate4(acc, afrag_h(aoth, 128, mrow, kt * 32 + ch), wb, ntb, 32 + kt, lane);
  // region 3: own controller state (hh term, ASZ = 128)
#pragma unroll
  for (int kt = 0; kt < 4; ++kt)
    gate4(acc, afrag_h(aself, 128, mrow, kt * 32 + ch), wb, ntb, 36 + kt, lane);

  int u = ntb * 16 + (lane & 15);
  float bi = gldf(bih + u)       + gldf(bhh + u);
  float bf = gldf(bih + 128 + u) + gldf(bhh + 128 + u);
  float bg = gldf(bih + 256 + u) + gldf(bhh + 256 + u);
  float bo = gldf(bih + 384 + u) + gldf(bhh + 384 + u);
  int mbase = mt * 16 + (lane >> 4) * 8;
#pragma unroll
  for (int r = 0; r < 8; ++r) {
    int m = mbase + r;
    float iv = sigm(acc[0][r] + bi);
    float fv = sigm(acc[1][r] + bf);
    float gv = tanhf(acc[2][r] + bg);
    float ov = sigm(acc[3][r] + bo);
    float c2 = fv * gldf(ahcs + m * 128 + u) + iv * gv;
    gstf(ahcd + m * 128 + u, c2);
    gsth(ahxd + m * 128 + u, (_Float16)(ov * tanhf(c2)));
  }
}

// ---------- phase: policy GEMM (64x128 @ 128x3072) + gating epilogue -------
static __device__ void pol_phase(int wave, int lane,
    const _Float16* ah, const _Float16* wb, const float* pb,
    const float* xrole, const float* hrole, float* polbuf, _Float16* uA) {
  for (int j = 0; j < 6; ++j) {          // 768 tiles over 128 waves
    int tile = wave + 128 * j;
    int mt = tile & 3, nt = tile >> 2;
    int mrow = mt * 16 + (lane & 15);
    int ch   = (lane >> 4) * 8;
    v8f acc = {};
#pragma unroll
    for (int kt = 0; kt < 4; ++kt)
      acc = wmma16(afrag_h(ah, 128, mrow, kt * 32 + ch),
                   bfrag(wb, nt * 4 + kt, lane), acc);
    int n = nt * 16 + (lane & 15);
    int mbase = mt * 16 + (lane >> 4) * 8;
    float pbn = gldf(pb + n);
#pragma unroll
    for (int r = 0; r < 8; ++r) {
      int m = mbase + r;
      float p = acc[r] + pbn;
      if (n < 512)
        gsth(uA + (size_t)m * 1024 + n, (_Float16)(gldf(xrole + m * 512 + n) * p));
      else if (n < 1024)
        gsth(uA + (size_t)m * 1024 + n,
             (_Float16)(gldf(hrole + m * 512 + n - 512) * p));
      else
        gstf(polbuf + (size_t)m * 3072 + n, p);
    }
  }
}

// ---------- phase: aRNN two GEMMs (K=512 each) + tanh epilogue -------------
static __device__ void arnn_phase(int wave, int lane,
    const _Float16* uA, const _Float16* wb, const float* polbuf, const float* fb,
    float* fhxd, _Float16* fhxhd, float* outp) {
  int mt = wave >> 5, nt = wave & 31;    // 128 tiles over 128 waves
  int mrow = mt * 16 + (lane & 15);
  int ch   = (lane >> 4) * 8;
  v8f acc1 = {}, acc2 = {};
#pragma unroll 2
  for (int kt = 0; kt < 16; ++kt)        // (x*p_in) @ fW_ih.T
    acc1 = wmma16(afrag_h(uA, 1024, mrow, kt * 32 + ch),
                  bfrag(wb, nt * 32 + kt, lane), acc1);
#pragma unroll 2
  for (int kt = 16; kt < 32; ++kt)       // (h*p_h) @ fW_hh.T
    acc2 = wmma16(afrag_h(uA, 1024, mrow, kt * 32 + ch),
                  bfrag(wb, nt * 32 + kt, lane), acc2);
  int n = nt * 16 + (lane & 15);
  int mbase = mt * 16 + (lane >> 4) * 8;
  float bn = gldf(fb + n);
#pragma unroll
  for (int r = 0; r < 8; ++r) {
    int m = mbase + r;
    const float* pr = polbuf + (size_t)m * 3072;
    float v = tanhf(acc1[r] * gldf(pr + 1024 + n) + bn * gldf(pr + 2048 + n) +
                    acc2[r] * gldf(pr + 1536 + n));
    gstf(fhxd + m * 512 + n, v);
    gsth(fhxhd + m * 512 + n, (_Float16)v);
    if (outp) gstf(outp + m * 512 + n, v);
  }
}

// ---------- persistent main kernel ----------
__global__ __launch_bounds__(256) void arnn_main(Args A) {
  int wave = ((blockIdx.x * blockDim.x) + threadIdx.x) >> 5;   // 0..127
  int lane = threadIdx.x & 31;
  const _Float16* wL0 = A.wfrag;
  const _Float16* wP0 = A.wfrag + (size_t)FR_LSTM * 512;
  const _Float16* wF0 = A.wfrag + (size_t)(FR_LSTM + FR_POL) * 512;
  const _Float16* wL1 = A.wfrag + (size_t)FR_PERL * 512;
  const _Float16* wP1 = wL1 + (size_t)FR_LSTM * 512;
  const _Float16* wF1 = wL1 + (size_t)(FR_LSTM + FR_POL) * 512;

  for (int t = 0; t < TT; ++t) {
    int s = t & 1, d = s ^ 1;            // state double-buffer slots
    const float* xt = A.x + (size_t)t * BB * 512;
    const _Float16 *ahx0s = A.ahx0h + s * 8192,  *ahx1s = A.ahx1h + s * 8192;
    _Float16       *ahx0d = A.ahx0h + d * 8192,  *ahx1d = A.ahx1h + d * 8192;
    const _Float16 *fhx0s = A.fhx0h + s * 32768, *fhx1s = A.fhx1h + s * 32768;
    _Float16       *fhx0d = A.fhx0h + d * 32768, *fhx1d = A.fhx1h + d * 32768;

    // Layer 0: A = [x(t) | fhx0(s) | ahx1(s) | ahx0(s)]
    lstm_phase(wave, lane, xt, nullptr, fhx0s, ahx1s, ahx0s, wL0,
               A.bih0, A.bhh0, A.ahc0 + s * 8192, A.ahc0 + d * 8192, ahx0d);
    grid_barrier(A.bar, A.bar + 1);
    pol_phase(wave, lane, ahx0d, wP0, A.pb0, xt, A.fhx0f + s * 32768,
              A.polbuf, A.uA);
    grid_barrier(A.bar, A.bar + 1);
    arnn_phase(wave, lane, A.uA, wF0, A.polbuf, A.fb0,
               A.fhx0f + d * 32768, fhx0d, nullptr);
    grid_barrier(A.bar, A.bar + 1);

    // Layer 1: A = [fhx0(new) | fhx1(s) | ahx0(new) | ahx1(s)]
    lstm_phase(wave, lane, nullptr, fhx0d, fhx1s, ahx0d, ahx1s, wL1,
               A.bih1, A.bhh1, A.ahc1 + s * 8192, A.ahc1 + d * 8192, ahx1d);
    grid_barrier(A.bar, A.bar + 1);
    pol_phase(wave, lane, ahx1d, wP1, A.pb1, A.fhx0f + d * 32768,
              A.fhx1f + s * 32768, A.polbuf, A.uA);
    grid_barrier(A.bar, A.bar + 1);
    arnn_phase(wave, lane, A.uA, wF1, A.polbuf, A.fb1,
               A.fhx1f + d * 32768, fhx1d, A.out + (size_t)t * BB * 512);
    grid_barrier(A.bar, A.bar + 1);
  }
}

// ---------- prep: swizzle all weights into WMMA B-fragment order (f16) -----
__global__ void prep_weights(const float* aWih0, const float* aWhh0,
                             const float* pW0, const float* fWih0, const float* fWhh0,
                             const float* aWih1, const float* aWhh1,
                             const float* pW1, const float* fWih1, const float* fWhh1,
                             _Float16* wfrag) {
  int idx = blockIdx.x * blockDim.x + threadIdx.x;
  const int PERL = FR_PERL * 512;
  if (idx >= 2 * PERL) return;
  int l = idx / PERL;
  int r = idx % PERL;
  int frag = r >> 9;
  int e    = r & 15;
  int ln   = (r >> 4) & 31;
  int k_off  = (ln >> 4) * 8 + (e < 8 ? e : e + 8);  // ISA 16-bit B layout
  int n_in16 = ln & 15;
  float v;
  if (frag < FR_LSTM) {                      // frag = nt*40 + kt, K = 1152|128
    int nt = frag / 40, kt = frag % 40;
    int n = nt * 16 + n_in16, k = kt * 32 + k_off;
    const float* Wih = l ? aWih1 : aWih0;
    const float* Whh = l ? aWhh1 : aWhh0;
    v = (k < 1152) ? gldf(Wih + (size_t)n * 1152 + k)
                   : gldf(Whh + (size_t)n * 128 + (k - 1152));
  } else if (frag < FR_LSTM + FR_POL) {      // frag-1280 = nt*4 + kt
    int f2 = frag - FR_LSTM;
    int nt = f2 >> 2, kt = f2 & 3;
    int n = nt * 16 + n_in16, k = kt * 32 + k_off;
    const float* W = l ? pW1 : pW0;
    v = gldf(W + (size_t)n * 128 + k);
  } else {                                   // frag-2048 = nt*32 + kt, K = 512|512
    int f2 = frag - (FR_LSTM + FR_POL);
    int nt = f2 >> 5, kt = f2 & 31;
    int n = nt * 16 + n_in16, k = kt * 32 + k_off;
    const float* Wih = l ? fWih1 : fWih0;
    const float* Whh = l ? fWhh1 : fWhh0;
    v = (k < 512) ? gldf(Wih + (size_t)n * 512 + k)
                  : gldf(Whh + (size_t)n * 512 + (k - 512));
  }
  gsth(wfrag + idx, (_Float16)v);
}

__global__ void prep_zero(unsigned* p, int n) {
  int i = blockIdx.x * blockDim.x + threadIdx.x;
  if (i < n) *(GAS unsigned*)(p + i) = 0u;
}

// ---------- host launcher ----------
extern "C" void kernel_launch(void* const* d_in, const int* in_sizes, int n_in,
                              void* d_out, int out_size, void* d_ws, size_t ws_size,
                              hipStream_t stream) {
  (void)in_sizes; (void)n_in; (void)out_size; (void)ws_size;
  const float* x     = (const float*)d_in[0];
  const float* aWih0 = (const float*)d_in[1];
  const float* aWhh0 = (const float*)d_in[2];
  const float* bih0  = (const float*)d_in[3];
  const float* bhh0  = (const float*)d_in[4];
  const float* pW0   = (const float*)d_in[5];
  const float* pb0   = (const float*)d_in[6];
  const float* fWih0 = (const float*)d_in[7];
  const float* fWhh0 = (const float*)d_in[8];
  const float* fb0   = (const float*)d_in[9];
  const float* aWih1 = (const float*)d_in[10];
  const float* aWhh1 = (const float*)d_in[11];
  const float* bih1  = (const float*)d_in[12];
  const float* bhh1  = (const float*)d_in[13];
  const float* pW1   = (const float*)d_in[14];
  const float* pb1   = (const float*)d_in[15];
  const float* fWih1 = (const float*)d_in[16];
  const float* fWhh1 = (const float*)d_in[17];
  const float* fb1   = (const float*)d_in[18];

  char* ws = (char*)d_ws;
  size_t off = 0;
  _Float16* wfrag = (_Float16*)(ws + off); off += (size_t)2 * FR_PERL * 512 * 2; // 6,291,456
  char* stateBase = ws + off;
  _Float16* ahx0h = (_Float16*)(ws + off); off += 2 * 8192 * 2;
  _Float16* ahx1h = (_Float16*)(ws + off); off += 2 * 8192 * 2;
  _Float16* fhx0h = (_Float16*)(ws + off); off += 2 * 32768 * 2;
  _Float16* fhx1h = (_Float16*)(ws + off); off += 2 * 32768 * 2;
  float* ahc0  = (float*)(ws + off); off += 2 * 8192 * 4;
  float* ahc1  = (float*)(ws + off); off += 2 * 8192 * 4;
  float* fhx0f = (float*)(ws + off); off += 2 * 32768 * 4;
  float* fhx1f = (float*)(ws + off); off += 2 * 32768 * 4;
  unsigned* bar = (unsigned*)(ws + off); off += 256;
  size_t zeroWords = (size_t)(ws + off - stateBase) / 4;     // states + barrier
  float* polbuf = (float*)(ws + off); off += (size_t)BB * 3072 * 4;
  _Float16* uA = (_Float16*)(ws + off); off += (size_t)BB * 1024 * 2;

  // 1) swizzle weights to f16 fragments (runs inside graph each replay)
  int wtot = 2 * FR_PERL * 512;
  prep_weights<<<(wtot + 255) / 256, 256, 0, stream>>>(
      aWih0, aWhh0, pW0, fWih0, fWhh0, aWih1, aWhh1, pW1, fWih1, fWhh1, wfrag);
  // 2) zero recurrent state + barrier (required every call: ws is not reset)
  prep_zero<<<((int)zeroWords + 255) / 256, 256, 0, stream>>>((unsigned*)stateBase,
                                                              (int)zeroWords);
  // 3) persistent scan kernel
  Args A;
  A.x = x;
  A.bih0 = bih0; A.bhh0 = bhh0; A.pb0 = pb0; A.fb0 = fb0;
  A.bih1 = bih1; A.bhh1 = bhh1; A.pb1 = pb1; A.fb1 = fb1;
  A.wfrag = wfrag;
  A.ahx0h = ahx0h; A.ahx1h = ahx1h; A.fhx0h = fhx0h; A.fhx1h = fhx1h;
  A.ahc0 = ahc0; A.ahc1 = ahc1; A.fhx0f = fhx0f; A.fhx1f = fhx1f;
  A.bar = bar; A.polbuf = polbuf; A.uA = uA;
  A.out = (float*)d_out;
  arnn_main<<<NWG, 256, 0, stream>>>(A);
}
